// IlluminationGuideAttention_77996606095739
// MI455X (gfx1250) — matrix operand order repaired and checked by reading, not compiled, not run on hardware
//
#include <hip/hip_runtime.h>
#include <hip/hip_bf16.h>

#define DIMC  192
#define HEADS 4
#define CH    48
#define NB    8          // batch
#define HW    16384
#define OCH   576        // 3*DIMC
#define PK    224        // proj K padded (204 -> 224)
#define KSPL  8          // split-K factor for logits GEMM

typedef __attribute__((ext_vector_type(16))) __bf16 v16bf;
typedef __attribute__((ext_vector_type(8)))  __bf16 v8bf;
typedef __attribute__((ext_vector_type(4)))  __bf16 v4bf;
typedef __attribute__((ext_vector_type(8)))  float  v8f;
typedef __attribute__((ext_vector_type(4)))  float  vf4;

__device__ __forceinline__ v8f wmma_bf16(v16bf a, v16bf b, v8f c) {
  return __builtin_amdgcn_wmma_f32_16x16x32_bf16(false, a, false, b, (short)0, c, false, false);
}
__device__ __forceinline__ v16bf cat8(v8bf a, v8bf b) {
  return __builtin_shufflevector(a, b, 0,1,2,3,4,5,6,7,8,9,10,11,12,13,14,15);
}

// ---------------------------------------------------------------- prepack ---
__global__ void k_prepack(const float* __restrict__ qkv_w, const float* __restrict__ proj_w,
                          __bf16* __restrict__ wq, __bf16* __restrict__ wp) {
  int i = blockIdx.x * 256 + threadIdx.x;
  if (i < OCH * DIMC) wq[i] = (__bf16)qkv_w[i];
  int j = i - OCH * DIMC;
  if (j >= 0 && j < DIMC * PK) {
    int o = j / PK, k = j % PK;
    wp[j] = (__bf16)((k < DIMC + 12) ? proj_w[o * (DIMC + 12) + k] : 0.f);
  }
}

// ------------------------------------------------------------- qkv 1x1 GEMM -
__global__ void k_qkv(const float* __restrict__ x, const __bf16* __restrict__ wq,
                      __bf16* __restrict__ qkv_pre) {
  extern __shared__ char smem[];
  __bf16* xt = (__bf16*)smem;                        // [128][192]
  int b = blockIdx.y, n0 = blockIdx.x * 128;
  int tid = threadIdx.x, wave = tid >> 5, lane = tid & 31;
  int m = lane & 15, g = lane >> 4;

  for (int i = tid; i < DIMC * 32; i += 256) {       // 192 rows x 32 float4
    int c = i >> 5, n4 = (i & 31) * 4;
    vf4 v = *reinterpret_cast<const vf4*>(x + ((size_t)(b * DIMC + c)) * HW + n0 + n4);
    #pragma unroll
    for (int j = 0; j < 4; ++j) xt[(n4 + j) * DIMC + c] = (__bf16)v[j];
  }
  __syncthreads();

  for (int t = wave; t < 288; t += 8) {              // 36 m-tiles x 8 n-tiles
    int mt = t >> 3, nt = t & 7;
    const __bf16* arow = wq + (mt * 16 + m) * DIMC;
    v8f acc = {};
    #pragma unroll
    for (int ks = 0; ks < 6; ++ks) {                 // K = 192
      v8bf a0 = *reinterpret_cast<const v8bf*>(arow + ks * 32 + 8 * g);
      v8bf a1 = *reinterpret_cast<const v8bf*>(arow + ks * 32 + 16 + 8 * g);
      v16bf bb = cat8(*reinterpret_cast<const v8bf*>(xt + (nt * 16 + m) * DIMC + ks * 32 + 8 * g),
                      *reinterpret_cast<const v8bf*>(xt + (nt * 16 + m) * DIMC + ks * 32 + 16 + 8 * g));
      acc = wmma_bf16(cat8(a0, a1), bb, acc);
    }
    int o0 = mt * 16 + 8 * g, n = n0 + nt * 16 + m;
    __bf16* dst = qkv_pre + ((size_t)(b * OCH + o0)) * HW + n;
    #pragma unroll
    for (int i = 0; i < 8; ++i) dst[(size_t)i * HW] = (__bf16)acc[i];
  }
}

// ------------------------------------------------- depthwise 3x3 (bf16 I/O) -
__global__ void k_dw(const __bf16* __restrict__ qkv_pre, const float* __restrict__ dw_w,
                     __bf16* __restrict__ qkv_dwb) {
  long idx = (long)blockIdx.x * 256 + threadIdx.x;   // b*576*HW total
  int n = idx & (HW - 1);
  int oc = (int)((idx >> 14) % OCH);
  int b  = (int)((idx >> 14) / OCH);
  int y = n >> 7, xx = n & 127;
  const float*  w   = dw_w + oc * 9;
  const __bf16* src = qkv_pre + (((size_t)(b * OCH + oc)) << 14);
  float s = 0.f;
  #pragma unroll
  for (int dy = -1; dy <= 1; ++dy) {
    int yy = y + dy;
    if (yy < 0 || yy > 127) continue;
    #pragma unroll
    for (int dx = -1; dx <= 1; ++dx) {
      int xc = xx + dx;
      if (xc < 0 || xc > 127) continue;
      s += w[(dy + 1) * 3 + (dx + 1)] * (float)src[yy * 128 + xc];
    }
  }
  qkv_dwb[idx] = (__bf16)s;
}

// ------------------------------------------------------------ svp 1x1 (3->12)
__global__ void k_svp(const float* __restrict__ svp_fea, const float* __restrict__ svp_w,
                      __bf16* __restrict__ svp_qb) {
  int idx = blockIdx.x * 256 + threadIdx.x;          // b*HW
  int b = idx >> 14, n = idx & (HW - 1);
  float f0 = svp_fea[(((size_t)(b * 3 + 0)) << 14) + n];
  float f1 = svp_fea[(((size_t)(b * 3 + 1)) << 14) + n];
  float f2 = svp_fea[(((size_t)(b * 3 + 2)) << 14) + n];
  #pragma unroll
  for (int o = 0; o < 12; ++o)
    svp_qb[(((size_t)(b * 12 + o)) << 14) + n] =
        (__bf16)(svp_w[o * 3] * f0 + svp_w[o * 3 + 1] * f1 + svp_w[o * 3 + 2] * f2);
}

// -------------------------------------------- inverse L2 norms over n=16384 -
__global__ void k_norm(const __bf16* __restrict__ qkv_dwb, const __bf16* __restrict__ svp_qb,
                       float* __restrict__ invn) {
  __shared__ float red[256];
  int rid = blockIdx.x, tid = threadIdx.x;
  const __bf16* p;
  if (rid < NB * 384) {
    int b = rid / 384, ch = rid % 384;
    p = qkv_dwb + ((size_t)(b * OCH + ch)) * HW;
  } else {
    p = svp_qb + ((size_t)(rid - NB * 384)) * HW;
  }
  float s = 0.f;
  for (int i = tid * 8; i < HW; i += 256 * 8) {
    v8bf v = *reinterpret_cast<const v8bf*>(p + i);
    #pragma unroll
    for (int j = 0; j < 8; ++j) { float f = (float)v[j]; s += f * f; }
  }
  red[tid] = s; __syncthreads();
  for (int off = 128; off > 0; off >>= 1) {
    if (tid < off) red[tid] += red[tid + off];
    __syncthreads();
  }
  if (tid == 0) invn[rid] = 1.f / fmaxf(sqrtf(red[0]), 1e-12f);
}

// ------------------------------------------ logits QK^T, split-K partials ---
// Block = (b,h,s): partial[c,d] = sum_{n in slice s} q[c,n]*k[d,n].
// 12 waves = 12 tiles (4 mt x 3 nt); pure bf16 loads + WMMA, K-slice = 2048.
__global__ void k_logits_part(const __bf16* __restrict__ qkv_dwb,
                              const __bf16* __restrict__ svp_qb,
                              float* __restrict__ pbuf) {
  int blk = blockIdx.x;                              // bh*KSPL + s
  int bh = blk >> 3, s = blk & (KSPL - 1);
  int b = bh >> 2, h = bh & 3;
  int tid = threadIdx.x, wave = tid >> 5, lane = tid & 31;
  int mt = wave & 3, nt = wave >> 2;                 // 4 x 3 tiles
  int m = lane & 15, g = lane >> 4;

  int cm = mt * 16 + m;                              // query row (incl. svp rows)
  const __bf16* pa;
  if (cm < CH)           pa = qkv_dwb + ((size_t)(b * OCH + h * CH + cm)) * HW;
  else if (cm < CH + 3)  pa = svp_qb  + ((size_t)(b * 12 + h * 3 + (cm - CH))) * HW;
  else                   pa = qkv_dwb;               // pad rows: zeroed by scale later
  int d = nt * 16 + m;                               // key row
  const __bf16* pb = qkv_dwb + ((size_t)(b * OCH + DIMC + h * CH + d)) * HW;

  int nbeg = s * (HW / KSPL);
  v8f acc = {};
  #pragma unroll 4
  for (int n0 = nbeg; n0 < nbeg + HW / KSPL; n0 += 32) {
    __builtin_prefetch(pa + n0 + 512, 0, 0);
    __builtin_prefetch(pb + n0 + 512, 0, 0);
    v8bf a0 = *reinterpret_cast<const v8bf*>(pa + n0 + 8 * g);
    v8bf a1 = *reinterpret_cast<const v8bf*>(pa + n0 + 16 + 8 * g);
    v16bf bb = cat8(*reinterpret_cast<const v8bf*>(pb + n0 + 8 * g),
                    *reinterpret_cast<const v8bf*>(pb + n0 + 16 + 8 * g));
    acc = wmma_bf16(cat8(a0, a1), bb, acc);
  }

  float* dst = pbuf + ((size_t)blk * 64 + mt * 16 + 8 * g) * 48 + nt * 16 + m;
  #pragma unroll
  for (int i = 0; i < 8; ++i) dst[i * 48] = acc[i];
}

// -------------------------- reduce split-K partials, scale, softmax -> attn -
__global__ void k_softmax(const float* __restrict__ pbuf, const float* __restrict__ invn,
                          const float* __restrict__ temperature, __bf16* __restrict__ attn) {
  __shared__ float ll[64 * 48];
  int bh = blockIdx.x, b = bh >> 2, h = bh & 3;
  int tid = threadIdx.x;
  float tmp = temperature[h];

  for (int i = tid; i < 64 * 48; i += 256) {
    int r = i / 48, d = i % 48;
    float s = 0.f;
    #pragma unroll
    for (int ss = 0; ss < KSPL; ++ss)
      s += pbuf[((size_t)(bh * KSPL + ss) * 64 + r) * 48 + d];
    float sa;
    if (r < CH)          sa = invn[b * 384 + h * CH + r];
    else if (r < CH + 3) sa = invn[NB * 384 + b * 12 + h * 3 + (r - CH)];
    else                 sa = 0.f;
    float sb = invn[b * 384 + DIMC + h * CH + d];
    ll[i] = s * sa * sb * tmp;
  }
  __syncthreads();

  if (tid < 64) {                                    // softmax over d (48)
    int r = tid;
    __bf16* arow = attn + ((size_t)bh) * 64 * 64 + r * 64;
    if (r < CH + 3) {
      float mx = -3.4e38f;
      for (int dd = 0; dd < 48; ++dd) mx = fmaxf(mx, ll[r * 48 + dd]);
      float s = 0.f;
      for (int dd = 0; dd < 48; ++dd) s += __expf(ll[r * 48 + dd] - mx);
      float is = 1.f / s;
      for (int dd = 0; dd < 48; ++dd) arow[dd] = (__bf16)(__expf(ll[r * 48 + dd] - mx) * is);
      for (int dd = 48; dd < 64; ++dd) arow[dd] = (__bf16)0.f;
    } else {
      for (int dd = 0; dd < 64; ++dd) arow[dd] = (__bf16)0.f;
    }
  }
}

// --------------------------------- fused attn@V (per head) + 1x1 projection -
__global__ void k_out(const __bf16* __restrict__ qkv_dwb, const __bf16* __restrict__ attn,
                      const __bf16* __restrict__ wp, float* __restrict__ out) {
  extern __shared__ char smem[];
  __bf16* vt  = (__bf16*)smem;                       // [64][64]
  __bf16* mid = (__bf16*)(smem + 64 * 64 * sizeof(__bf16)); // [64][224]
  int b = blockIdx.y, n0 = blockIdx.x * 64;
  int tid = threadIdx.x, wave = tid >> 5, lane = tid & 31;
  int m = lane & 15, g = lane >> 4;

  for (int i = tid; i < 64 * 20; i += 384) {         // zero proj-K pad 204..223
    int n = i / 20, ch = 204 + i % 20;
    mid[n * PK + ch] = (__bf16)0.f;
  }

  for (int h = 0; h < HEADS; ++h) {
    __syncthreads();
    for (int i = tid; i < 64 * 16; i += 384) {       // zero d pad 48..63
      int n = i >> 4, dd = 48 + (i & 15);
      vt[n * 64 + dd] = (__bf16)0.f;
    }
    for (int i = tid; i < 64 * 12; i += 384) {       // stage v transposed (4-wide)
      int dd = i >> 4, n4 = (i & 15) * 4;
      v4bf v = *reinterpret_cast<const v4bf*>(
          qkv_dwb + ((size_t)(b * OCH + 2 * DIMC + h * CH + dd)) * HW + n0 + n4);
      #pragma unroll
      for (int j = 0; j < 4; ++j) vt[(n4 + j) * 64 + dd] = v[j];
    }
    __syncthreads();

    const __bf16* ab = attn + ((size_t)(b * HEADS + h)) * 64 * 64;
    for (int t = wave; t < 16; t += 12) {            // 4 mt x 4 nt tiles
      int mt = t >> 2, nt = t & 3;
      v8f acc = {};
      #pragma unroll
      for (int ks = 0; ks < 2; ++ks) {               // K = 64 (48 + zero pad)
        v8bf a0 = *reinterpret_cast<const v8bf*>(ab + (mt * 16 + m) * 64 + ks * 32 + 8 * g);
        v8bf a1 = *reinterpret_cast<const v8bf*>(ab + (mt * 16 + m) * 64 + ks * 32 + 16 + 8 * g);
        v16bf bb = cat8(*reinterpret_cast<const v8bf*>(vt + (nt * 16 + m) * 64 + ks * 32 + 8 * g),
                        *reinterpret_cast<const v8bf*>(vt + (nt * 16 + m) * 64 + ks * 32 + 16 + 8 * g));
        acc = wmma_bf16(cat8(a0, a1), bb, acc);
      }
      int nl = nt * 16 + m, c0 = mt * 16 + 8 * g;
      #pragma unroll
      for (int i = 0; i < 8; ++i) {
        int c = c0 + i;
        if (c < CH + 3) mid[nl * PK + h * (CH + 3) + c] = (__bf16)acc[i];
      }
    }
  }
  __syncthreads();

  for (int t = wave; t < 48; t += 12) {              // proj: 12 mt x 4 nt tiles
    int mt = t / 4, nt = t % 4;
    const __bf16* arow = wp + (mt * 16 + m) * PK;
    v8f acc = {};
    #pragma unroll
    for (int ks = 0; ks < 7; ++ks) {                 // K = 224 (204 + pad)
      v8bf a0 = *reinterpret_cast<const v8bf*>(arow + ks * 32 + 8 * g);
      v8bf a1 = *reinterpret_cast<const v8bf*>(arow + ks * 32 + 16 + 8 * g);
      v16bf bb = cat8(*reinterpret_cast<const v8bf*>(mid + (nt * 16 + m) * PK + ks * 32 + 8 * g),
                      *reinterpret_cast<const v8bf*>(mid + (nt * 16 + m) * PK + ks * 32 + 16 + 8 * g));
      acc = wmma_bf16(cat8(a0, a1), bb, acc);
    }
    int o0 = mt * 16 + 8 * g, n = n0 + nt * 16 + m;
    float* dst = out + ((size_t)(b * DIMC + o0)) * HW + n;
    #pragma unroll
    for (int i = 0; i < 8; ++i) dst[(size_t)i * HW] = acc[i];
  }
}

// ---------------------------------------------------------------------------
extern "C" void kernel_launch(void* const* d_in, const int* in_sizes, int n_in,
                              void* d_out, int out_size, void* d_ws, size_t ws_size,
                              hipStream_t stream) {
  const float* x           = (const float*)d_in[0];
  const float* svp_fea     = (const float*)d_in[1];
  const float* qkv_w       = (const float*)d_in[2];
  const float* dw_w        = (const float*)d_in[3];
  const float* svp_w       = (const float*)d_in[4];
  const float* proj_w      = (const float*)d_in[5];
  const float* temperature = (const float*)d_in[6];
  float* out = (float*)d_out;

  char* ws = (char*)d_ws;
  const size_t SZ_QKV = (size_t)NB * OCH * HW * sizeof(__bf16);      // 144 MB
  const size_t SZ_SVP = (size_t)NB * 12 * HW * sizeof(__bf16);       // 3 MB
  const size_t SZ_INV = 4096 * sizeof(float);
  const size_t SZ_ATT = (size_t)NB * HEADS * 64 * 64 * sizeof(__bf16);
  const size_t SZ_WQ  = (size_t)OCH * DIMC * sizeof(__bf16);
  const size_t SZ_WP  = (size_t)DIMC * PK * sizeof(__bf16);

  __bf16* qkv_pre = (__bf16*)ws;
  __bf16* qkv_dwb = (__bf16*)(ws + SZ_QKV);
  __bf16* svp_qb  = (__bf16*)(ws + 2 * SZ_QKV);
  float*  invn    = (float*)(ws + 2 * SZ_QKV + SZ_SVP);
  __bf16* attn    = (__bf16*)(ws + 2 * SZ_QKV + SZ_SVP + SZ_INV);
  __bf16* wq      = (__bf16*)(ws + 2 * SZ_QKV + SZ_SVP + SZ_INV + SZ_ATT);
  __bf16* wp      = (__bf16*)(ws + 2 * SZ_QKV + SZ_SVP + SZ_INV + SZ_ATT + SZ_WQ);
  float*  pbuf    = (float*)(ws + 2 * SZ_QKV + SZ_SVP + SZ_INV + SZ_ATT + SZ_WQ + SZ_WP);

  k_prepack<<<600, 256, 0, stream>>>(qkv_w, proj_w, wq, wp);
  k_qkv<<<dim3(HW / 128, NB), 256, 128 * DIMC * sizeof(__bf16), stream>>>(x, wq, qkv_pre);
  k_dw<<<(NB * OCH * HW) / 256, 256, 0, stream>>>(qkv_pre, dw_w, qkv_dwb);
  k_svp<<<(NB * HW) / 256, 256, 0, stream>>>(svp_fea, svp_w, svp_qb);
  k_norm<<<NB * (384 + 12), 256, 0, stream>>>(qkv_dwb, svp_qb, invn);
  k_logits_part<<<NB * HEADS * KSPL, 384, 0, stream>>>(qkv_dwb, svp_qb, pbuf);
  k_softmax<<<NB * HEADS, 256, 0, stream>>>(pbuf, invn, temperature, attn);
  k_out<<<dim3(HW / 64, NB), 384, (64 * 64 + 64 * PK) * sizeof(__bf16), stream>>>(qkv_dwb, attn, wp, out);
}
